// NewLLMResidual_83803401879983
// MI455X (gfx1250) — matrix operand order repaired and checked by reading, not compiled, not run on hardware
//
#include <hip/hip_runtime.h>
#include <hip/hip_bf16.h>

// ---------------------------------------------------------------------------
// CDNA5 / gfx1250 implementation of the NewLLMResidual reference.
//   tokens T = B*S = 4096, E = C = 512, H = 1024, V = 32000, NB = 3, NL = 2
// All GEMMs run on v_wmma_f32_16x16x32_bf16 (bf16 in, f32 accumulate).
// GEMM is software-pipelined (register-staged next tile overlaps WMMA phase)
// and the epilogue is staged through LDS for fully coalesced b128 stores.
// ---------------------------------------------------------------------------

typedef __attribute__((ext_vector_type(16))) __bf16 v16bf;
typedef __attribute__((ext_vector_type(8)))  float  v8f;

union Frag16 { v16bf v; unsigned int u[8]; };

#define TOKENS 4096
#define EDIM   512
#define HDIM   1024
#define VOCAB  32000
#define LNEPS  1e-5f

// bf16 conversion, round-to-nearest-even
__device__ __forceinline__ unsigned short f32_to_bf16(float f) {
    unsigned int u = __float_as_uint(f);
    u += 0x7FFFu + ((u >> 16) & 1u);
    return (unsigned short)(u >> 16);
}

// ---------------------------------------------------------------------------
// f32 -> bf16 bulk conversion (n must be a multiple of 4; true for our uses)
// ---------------------------------------------------------------------------
__global__ __launch_bounds__(256)
void cvt_f32_bf16_kernel(const float* __restrict__ src,
                         unsigned short* __restrict__ dst, long long n4) {
    long long i = ((long long)blockIdx.x * 256 + threadIdx.x);
    if (i >= n4) return;
    long long e = i * 4;
    float4 v = *(const float4*)(src + e);
    unsigned short p[4] = { f32_to_bf16(v.x), f32_to_bf16(v.y),
                            f32_to_bf16(v.z), f32_to_bf16(v.w) };
    *(uint2*)(dst + e) = *(const uint2*)p;
}

// ---------------------------------------------------------------------------
// shared-memory tree reduction over 256 threads, returns total to all threads
// ---------------------------------------------------------------------------
__device__ __forceinline__ float block_sum(float v, float* red, int tid) {
    red[tid] = v;
    __syncthreads();
    #pragma unroll
    for (int s = 128; s > 0; s >>= 1) {
        if (tid < s) red[tid] += red[tid + s];
        __syncthreads();
    }
    float r = red[0];
    __syncthreads();
    return r;
}

// ---------------------------------------------------------------------------
// Embedding gather + LayerNorm. One block (256 threads) per token.
// Writes: tok_f32 [T,512], ctx_f32 [T,512] (zeros),
//         act bf16 [T,1024] = concat(ctx=0, tok)
// ---------------------------------------------------------------------------
__global__ __launch_bounds__(256)
void embed_ln_kernel(const int* __restrict__ ids,
                     const float* __restrict__ table,
                     const float* __restrict__ g,
                     const float* __restrict__ b,
                     float* __restrict__ tok_f32,
                     float* __restrict__ ctx_f32,
                     unsigned short* __restrict__ act) {
    __shared__ float red[256];
    const int t   = blockIdx.x;
    const int tid = threadIdx.x;
    const float* src = table + (size_t)ids[t] * EDIM;
    float x0 = src[tid], x1 = src[tid + 256];

    float mean = block_sum(x0 + x1, red, tid) * (1.0f / EDIM);
    float d0 = x0 - mean, d1 = x1 - mean;
    float var = block_sum(d0 * d0 + d1 * d1, red, tid) * (1.0f / EDIM);
    float rstd = rsqrtf(var + LNEPS);

    float y0 = d0 * rstd * g[tid]       + b[tid];
    float y1 = d1 * rstd * g[tid + 256] + b[tid + 256];

    size_t tb5 = (size_t)t * EDIM;
    size_t tb1 = (size_t)t * HDIM;
    tok_f32[tb5 + tid]       = y0;
    tok_f32[tb5 + tid + 256] = y1;
    ctx_f32[tb5 + tid]       = 0.0f;
    ctx_f32[tb5 + tid + 256] = 0.0f;
    act[tb1 + tid]           = 0;                 // ctx half = 0 (bf16)
    act[tb1 + tid + 256]     = 0;
    act[tb1 + EDIM + tid]       = f32_to_bf16(y0); // tok half
    act[tb1 + EDIM + tid + 256] = f32_to_bf16(y1);
}

// ---------------------------------------------------------------------------
// Residual + LayerNorm on both halves. One block per token.
//   ctx = LN(ctx + h[:,0:512]; cg,cb),  tok = LN(tok + h[:,512:1024]; tg,tb)
// Also rewrites the bf16 concat activation buffer for the next block's GEMM.
// ---------------------------------------------------------------------------
__global__ __launch_bounds__(256)
void res_ln_kernel(const float* __restrict__ h,
                   const float* __restrict__ cg, const float* __restrict__ cb,
                   const float* __restrict__ tg, const float* __restrict__ tb,
                   float* __restrict__ ctx, float* __restrict__ tok,
                   unsigned short* __restrict__ act) {
    __shared__ float red[256];
    const int t   = blockIdx.x;
    const int tid = threadIdx.x;
    size_t tb5 = (size_t)t * EDIM;
    size_t tb1 = (size_t)t * HDIM;

    // ---- ctx half ----
    {
        float c0 = ctx[tb5 + tid]       + h[tb1 + tid];
        float c1 = ctx[tb5 + tid + 256] + h[tb1 + tid + 256];
        float mean = block_sum(c0 + c1, red, tid) * (1.0f / EDIM);
        float d0 = c0 - mean, d1 = c1 - mean;
        float var = block_sum(d0 * d0 + d1 * d1, red, tid) * (1.0f / EDIM);
        float rstd = rsqrtf(var + LNEPS);
        float y0 = d0 * rstd * cg[tid]       + cb[tid];
        float y1 = d1 * rstd * cg[tid + 256] + cb[tid + 256];
        ctx[tb5 + tid]       = y0;
        ctx[tb5 + tid + 256] = y1;
        act[tb1 + tid]       = f32_to_bf16(y0);
        act[tb1 + tid + 256] = f32_to_bf16(y1);
    }
    // ---- tok half ----
    {
        float c0 = tok[tb5 + tid]       + h[tb1 + EDIM + tid];
        float c1 = tok[tb5 + tid + 256] + h[tb1 + EDIM + tid + 256];
        float mean = block_sum(c0 + c1, red, tid) * (1.0f / EDIM);
        float d0 = c0 - mean, d1 = c1 - mean;
        float var = block_sum(d0 * d0 + d1 * d1, red, tid) * (1.0f / EDIM);
        float rstd = rsqrtf(var + LNEPS);
        float y0 = d0 * rstd * tg[tid]       + tb[tid];
        float y1 = d1 * rstd * tg[tid + 256] + tb[tid + 256];
        tok[tb5 + tid]       = y0;
        tok[tb5 + tid + 256] = y1;
        act[tb1 + EDIM + tid]       = f32_to_bf16(y0);
        act[tb1 + EDIM + tid + 256] = f32_to_bf16(y1);
    }
}

// ---------------------------------------------------------------------------
// WMMA GEMM:  D[M,N] = act(A[M,K(lda)] x W[N,K]^T + bias)
//   A, W bf16 row-major; accumulate f32 (v_wmma_f32_16x16x32_bf16).
//   Block: 256 thr = 8 waves (4 Mwaves x 2 Nwaves); block tile 64x128, K-step 32.
//   Each wave: 16(M) x 64(N) = 4 accumulators, A fragment reused 4x.
//   Software pipeline: next K-tile is loaded into registers while the current
//   tile's WMMAs execute; all 5 fragments are ds-loaded before the 4 WMMAs so
//   they issue back-to-back on the matrix pipe.
//   Epilogue stages the block tile through LDS (padded stride) and writes
//   fully-coalesced b128 global stores (bias + optional ReLU fused).
// ---------------------------------------------------------------------------
#define SAPAD 40                  // LDS tile row stride (elems): 80B rows
#define OPAD  132                 // epilogue f32 row stride (dwords): 528B rows

template <bool RELU, bool OUT_BF16>
__global__ __launch_bounds__(256)
void wmma_gemm_kernel(const unsigned short* __restrict__ A, int lda,
                      const unsigned short* __restrict__ W,
                      const float* __restrict__ bias,
                      void* __restrict__ Dout, int ldd,
                      int M, int N, int K) {
    // 64*SAPAD*2 (5120B, A) + 128*SAPAD*2 (10240B, B)  <=  64*OPAD*4 (33792B)
    __shared__ __align__(16) char smem[64 * OPAD * 4];
    unsigned short* sA   = (unsigned short*)smem;
    unsigned short* sB   = (unsigned short*)(smem + 64 * SAPAD * 2);
    float*          sOut = (float*)smem;

    const int tid  = threadIdx.x;
    const int lane = tid & 31;
    const int wave = tid >> 5;
    const int wm   = wave >> 1;   // 0..3  (M sub-tile)
    const int wn   = wave & 1;    // 0..1  (N sub-tile group)
    const int m0   = blockIdx.y * 64;
    const int n0   = blockIdx.x * 128;

    // cooperative tile-load coordinates
    const int arow = tid >> 2;          // 0..63
    const int acol = (tid & 3) << 3;    // 0,8,16,24  (8 bf16 = 16B each)
    const int brow = tid >> 1;          // 0..127
    const int bcol = (tid & 1) << 4;    // 0,16       (16 bf16 = 32B each)

    const unsigned short* Aptr = A + (size_t)(m0 + arow) * lda + acol;
    const unsigned short* Wptr = W + (size_t)(n0 + brow) * K + bcol;

    const v8f zero = {0.f, 0.f, 0.f, 0.f, 0.f, 0.f, 0.f, 0.f};
    v8f acc[4] = {zero, zero, zero, zero};

    const int r  = lane & 15;   // A: row M, B: col N, D: col N within sub-tile
    const int hi = lane >> 4;   // lane-half selector

    // ---- software pipeline prologue: tile 0 into registers ----
    uint4 av  = *(const uint4*)Aptr;
    uint4 bv0 = *(const uint4*)Wptr;
    uint4 bv1 = *(const uint4*)(Wptr + 8);

    const int ksteps = K >> 5;
    for (int ks = 0; ks < ksteps; ++ks) {
        // commit staged tile to LDS
        *(uint4*)(&sA[arow * SAPAD + acol])     = av;
        *(uint4*)(&sB[brow * SAPAD + bcol])     = bv0;
        *(uint4*)(&sB[brow * SAPAD + bcol + 8]) = bv1;
        __syncthreads();

        // issue next tile's global loads; they overlap the WMMA phase below
        if (ks + 1 < ksteps) {
            Aptr += 32;
            Wptr += 32;
            av  = *(const uint4*)Aptr;
            bv0 = *(const uint4*)Wptr;
            bv1 = *(const uint4*)(Wptr + 8);
            __builtin_prefetch(Aptr + 32, 0, 1);   // global_prefetch_b8, tile+2
            __builtin_prefetch(Wptr + 32, 0, 1);
        }

        // A fragment: 16x32 bf16, M = lane&15, K striped per ISA 7.12.2
        Frag16 af;
        const unsigned int* sArow = (const unsigned int*)&sA[(wm * 16 + r) * SAPAD];
        #pragma unroll
        for (int d = 0; d < 4; ++d) af.u[d]     = sArow[hi * 4 + d];      // K = hi*8 + 2d
        #pragma unroll
        for (int d = 0; d < 4; ++d) af.u[4 + d] = sArow[8 + hi * 4 + d];  // K = 16 + hi*8 + 2d

        // all 4 B fragments first (single dscnt wait), then 4 back-to-back WMMAs
        Frag16 bf[4];
        #pragma unroll
        for (int j = 0; j < 4; ++j) {
            const unsigned int* sBrow =
                (const unsigned int*)&sB[(wn * 64 + j * 16 + r) * SAPAD];
            #pragma unroll
            for (int d = 0; d < 4; ++d) bf[j].u[d] = sBrow[hi * 4 + d];
            #pragma unroll
            for (int d = 0; d < 4; ++d) bf[j].u[4 + d] = sBrow[8 + hi * 4 + d];
        }
        #pragma unroll
        for (int j = 0; j < 4; ++j)
            acc[j] = __builtin_amdgcn_wmma_f32_16x16x32_bf16(
                false, af.v, false, bf[j].v, (short)0, acc[j], false, false);

        __syncthreads();
    }

    // ---- epilogue: stage block tile in LDS (bias+ReLU fused), then b128 ----
    // D layout: VGPR rr -> M = wm*16 + hi*8 + rr ; N = lane&15 within sub-tile
    const int mrow0 = wm * 16 + hi * 8;
    #pragma unroll
    for (int j = 0; j < 4; ++j) {
        const int nl = wn * 64 + j * 16 + r;          // 0..127 within block tile
        const float bv = bias ? bias[n0 + nl] : 0.0f;
        #pragma unroll
        for (int rr = 0; rr < 8; ++rr) {
            float v = acc[j][rr] + bv;
            if (RELU) v = fmaxf(v, 0.0f);
            sOut[(mrow0 + rr) * OPAD + nl] = v;
        }
    }
    __syncthreads();

    if (OUT_BF16) {
        // 64 rows x 16 chunks of 8 bf16 (16B) = 1024 b128 stores, 4 per thread
        unsigned short* D = (unsigned short*)Dout;
        #pragma unroll
        for (int k = 0; k < 4; ++k) {
            int g    = k * 256 + tid;
            int row  = g >> 4;           // 0..63
            int cq   = g & 15;           // 0..15 (chunk of 8 elems)
            const float* s = &sOut[row * OPAD + cq * 8];
            unsigned short p[8];
            #pragma unroll
            for (int e = 0; e < 8; ++e) p[e] = f32_to_bf16(s[e]);
            *(uint4*)(&D[(size_t)(m0 + row) * ldd + n0 + cq * 8]) = *(const uint4*)p;
        }
    } else {
        // 64 rows x 32 chunks of 4 f32 (16B) = 2048 b128 stores, 8 per thread
        float* D = (float*)Dout;
        #pragma unroll
        for (int k = 0; k < 8; ++k) {
            int g   = k * 256 + tid;
            int row = g >> 5;            // 0..63
            int cq  = g & 31;            // 0..31 (chunk of 4 elems)
            uint4 v = *(const uint4*)&sOut[row * OPAD + cq * 4];
            *(uint4*)(&D[(size_t)(m0 + row) * ldd + n0 + cq * 4]) = v;
        }
    }
    (void)M; (void)N;
}

// ---------------------------------------------------------------------------
// Host-side orchestration
// ---------------------------------------------------------------------------
extern "C" void kernel_launch(void* const* d_in, const int* in_sizes, int n_in,
                              void* d_out, int out_size, void* d_ws, size_t ws_size,
                              hipStream_t stream) {
    (void)in_sizes; (void)n_in; (void)out_size; (void)ws_size;

    const int*   ids    = (const int*)  d_in[0];   // [4,1024]
    const float* table  = (const float*)d_in[1];   // [32000,512]
    const float* en_g   = (const float*)d_in[2];
    const float* en_b   = (const float*)d_in[3];
    const float* Wb     = (const float*)d_in[4];   // [3,2,1024,1024]
    const float* bb     = (const float*)d_in[5];   // [3,2,1024]
    const float* cn_g   = (const float*)d_in[6];   // [3,512]
    const float* cn_b   = (const float*)d_in[7];
    const float* tn_g   = (const float*)d_in[8];   // [3,512]
    const float* tn_b   = (const float*)d_in[9];
    const float* out_W  = (const float*)d_in[10];  // [32000,512]
    const float* out_b  = (const float*)d_in[11];
    float*       logits = (float*)d_out;           // [4,1024,32000]

    // workspace layout (all offsets 4KB aligned), total ~96 MB
    char* ws = (char*)d_ws;
    float*          tok_f32 = (float*)         (ws + 0);          //  8 MB
    float*          ctx_f32 = (float*)         (ws + 8388608);    //  8 MB
    float*          h_f32   = (float*)         (ws + 16777216);   // 16 MB
    unsigned short* act0    = (unsigned short*)(ws + 33554432);   //  8 MB concat(ctx,tok)
    unsigned short* act1    = (unsigned short*)(ws + 41943040);   //  8 MB layer-1 out
    unsigned short* Wb16    = (unsigned short*)(ws + 50331648);   // 12 MB
    unsigned short* oW16    = (unsigned short*)(ws + 62914560);   // 32 MB

    // 1) one-time weight conversion f32 -> bf16 (RNE)
    {
        long long nWb = 3LL * 2 * HDIM * HDIM;      // 6,291,456
        long long nOW = (long long)VOCAB * EDIM;    // 16,384,000
        hipLaunchKernelGGL(cvt_f32_bf16_kernel,
                           dim3((unsigned)((nWb / 4 + 255) / 256)), dim3(256), 0, stream,
                           Wb, Wb16, nWb / 4);
        hipLaunchKernelGGL(cvt_f32_bf16_kernel,
                           dim3((unsigned)((nOW / 4 + 255) / 256)), dim3(256), 0, stream,
                           out_W, oW16, nOW / 4);
    }

    // 2) embedding gather + LN
    hipLaunchKernelGGL(embed_ln_kernel, dim3(TOKENS), dim3(256), 0, stream,
                       ids, table, en_g, en_b, tok_f32, ctx_f32, act0);

    // 3) three FFN residual blocks
    const dim3 gFFN(HDIM / 128, TOKENS / 64);   // (8, 64)
    for (int bi = 0; bi < 3; ++bi) {
        const unsigned short* W0 = Wb16 + (size_t)(bi * 2 + 0) * HDIM * HDIM;
        const unsigned short* W1 = Wb16 + (size_t)(bi * 2 + 1) * HDIM * HDIM;
        const float* b0 = bb + (size_t)(bi * 2 + 0) * HDIM;
        const float* b1 = bb + (size_t)(bi * 2 + 1) * HDIM;

        // h1 = relu(act0 @ W0^T + b0)  -> bf16
        hipLaunchKernelGGL((wmma_gemm_kernel<true, true>), gFFN, dim3(256), 0, stream,
                           act0, HDIM, W0, b0, (void*)act1, HDIM,
                           TOKENS, HDIM, HDIM);
        // h2 = relu(h1 @ W1^T + b1)    -> f32
        hipLaunchKernelGGL((wmma_gemm_kernel<true, false>), gFFN, dim3(256), 0, stream,
                           act1, HDIM, W1, b1, (void*)h_f32, HDIM,
                           TOKENS, HDIM, HDIM);
        // residual + LN on both halves, refresh act0
        hipLaunchKernelGGL(res_ln_kernel, dim3(TOKENS), dim3(256), 0, stream,
                           h_f32, cn_g + bi * EDIM, cn_b + bi * EDIM,
                           tn_g + bi * EDIM, tn_b + bi * EDIM,
                           ctx_f32, tok_f32, act0);
    }

    // 4) logits = tok @ out_W^T + out_b   (tok half of act0: offset 512, lda 1024)
    const dim3 gOut(VOCAB / 128, TOKENS / 64);  // (250, 64)
    hipLaunchKernelGGL((wmma_gemm_kernel<false, false>), gOut, dim3(256), 0, stream,
                       act0 + EDIM, HDIM, oW16, out_b, (void*)logits, VOCAB,
                       TOKENS, VOCAB, EDIM);
}
